// MS_SSIM_Loss_50216757625558
// MI455X (gfx1250) — matrix operand order repaired and checked by reading, not compiled, not run on hardware
//
#include <hip/hip_runtime.h>
#include <hip/hip_bf16.h>

typedef __attribute__((ext_vector_type(16))) __bf16 v16bf;
typedef __attribute__((ext_vector_type(8)))  float  v8f;

// 11-tap Gaussian, sigma=1.5, normalized, zero-padded so that index (16+k)
// is in-bounds for k in [-16, 31]; taps live at [16..26].
__device__ __constant__ float GWP[48] = {
    0.f, 0.f, 0.f, 0.f, 0.f, 0.f, 0.f, 0.f,
    0.f, 0.f, 0.f, 0.f, 0.f, 0.f, 0.f, 0.f,
    0.00102838f, 0.00759877f, 0.03600077f, 0.10936060f, 0.21300553f,
    0.26601172f, 0.21300553f, 0.10936060f, 0.03600077f, 0.00759877f,
    0.00102838f,
    0.f, 0.f, 0.f, 0.f, 0.f, 0.f, 0.f, 0.f, 0.f, 0.f,
    0.f, 0.f, 0.f, 0.f, 0.f, 0.f, 0.f, 0.f, 0.f, 0.f, 0.f};

#define C1_ 0.0001f   // (0.01*1.0)^2
#define C2_ 0.0009f   // (0.03*1.0)^2

// ---------------------------------------------------------------------------
// Fused per-level SSIM statistics kernel.
// Grid: (ceil(Wv/64), ceil(Hv/16), N*C). Block: 128 threads (4 waves).
// Each wave produces a 16x16 tile of ssim_map / cs_map and accumulates sums.
// Separable 11-tap blur done as banded-matrix GEMMs on the WMMA bf16 pipe.
// Band matrices are zero for K>=26, so the data operand may carry arbitrary
// *finite* values in the K-pad region -> all operand packing is branch-free.
// ---------------------------------------------------------------------------
__global__ __launch_bounds__(128)
void ssim_stats_kernel(const float* __restrict__ X, const float* __restrict__ Y,
                       float* __restrict__ accSsim, float* __restrict__ accCs,
                       int H, int W, int Hv, int Wv)
{
    __shared__ float sX[26][80];     // 26 x 80 input tile (cols 74..79 = K-pad)
    __shared__ float sY[26][80];
    __shared__ float sH[4][32 * 16]; // per-wave H staging, rows 26..31 = pad

    const int tid  = threadIdx.x;
    const int wid  = tid >> 5;
    const int lane = tid & 31;
    const int half = lane >> 4;
    const int ln   = lane & 15;

    const int  nc      = blockIdx.z;
    const int  rowBase = blockIdx.y * 16;
    const int  colBase = blockIdx.x * 64;
    const long plane   = (long)nc * H * W;
    const float* Xp = X + plane;
    const float* Yp = Y + plane;

    float* sHw = sH[wid];

    // zero the 6 pad rows of this wave's H staging region exactly once:
    // they are read by the vertical-pass B operand (times zero weights) and
    // must be finite, not stale LDS bits that could decode as NaN.
    for (int i = lane; i < 6 * 16; i += 32) sHw[26 * 16 + i] = 0.0f;

    // ---- cooperative stage of the 26x80 input tiles (edge-clamped) ----
    for (int i = tid; i < 26 * 80; i += 128) {
        int r = i / 80, c = i % 80;
        int gr = rowBase + r; if (gr > H - 1) gr = H - 1;
        int gc = colBase + c; if (gc > W - 1) gc = W - 1;
        sX[r][c] = Xp[(long)gr * W + gc];
        sY[r][c] = Yp[(long)gr * W + gc];
    }
    __syncthreads();

    // ---- constant banded weight operands (per-lane, branch-free) ----
    // Bw: horizontal band matrix in B layout: element e -> K = (e>>3)*16+half*8+(e&7), N=ln
    //     Bw[K][N] = w[K-N]  (zero outside the 11-tap band and for K>=26)
    // Aw: vertical band matrix in A layout: M = ln, element e -> K
    //     Aw[M][K] = w[K-M]
    v16bf Bw, Aw;
#pragma unroll
    for (int e = 0; e < 16; ++e) {
        int kb = ((e >> 3) * 16) + half * 8 + (e & 7) - ln; // in [-15, 31]
        Bw[e] = (__bf16)GWP[16 + kb];
        int K  = ((e >> 3) * 16) + half * 8 + (e & 7);
        Aw[e] = (__bf16)GWP[16 + K - ln];               // in-bounds always
    }

    const int wcol0 = wid * 16;

    // ---- pack A operands for horizontal pass: row groups 0..15 and 10..25 --
    v16bf Ax[2], Ay[2], Axx[2], Ayy[2], Axy[2];
#pragma unroll
    for (int g = 0; g < 2; ++g) {
        const int row = g * 10 + ln;   // A layout: M = ln (both lane halves)
#pragma unroll
        for (int e = 0; e < 16; ++e) {
            int K = ((e >> 3) * 16) + half * 8 + (e & 7);   // in [0, 31]
            float x = sX[row][wcol0 + K];   // K>=26: finite garbage * 0 weight
            float y = sY[row][wcol0 + K];
            Ax [g][e] = (__bf16)x;
            Ay [g][e] = (__bf16)y;
            Axx[g][e] = (__bf16)(x * x);
            Ayy[g][e] = (__bf16)(y * y);
            Axy[g][e] = (__bf16)(x * y);
        }
    }

    // ---- per-field separable blur: 2 WMMA (H) + LDS relayout + 1 WMMA (V) --
    auto conv_field = [&](const v16bf (&A)[2]) -> v8f {
        v8f z = {};
        v8f h0 = __builtin_amdgcn_wmma_f32_16x16x32_bf16(
            false, A[0], false, Bw, (short)0, z, false, false);
        v8f h1 = __builtin_amdgcn_wmma_f32_16x16x32_bf16(
            false, A[1], false, Bw, (short)0, z, false, false);
        // scatter H rows (f32) to this wave's LDS region: D layout M = r+8*half
#pragma unroll
        for (int r = 0; r < 8; ++r) {
            sHw[(r + 8 * half) * 16 + ln]      = h0[r];   // rows 0..15
            sHw[(10 + r + 8 * half) * 16 + ln] = h1[r];   // rows 10..25
        }
        asm volatile("s_wait_dscnt 0" ::: "memory");  // wave-local DS fence
        // gather into B layout for the vertical GEMM: K = (e>>3)*16+half*8+(e&7), N = ln
        v16bf Bh;
#pragma unroll
        for (int e = 0; e < 16; ++e) {
            int rr = ((e >> 3) * 16) + half * 8 + (e & 7); // in [0, 31]
            Bh[e] = (__bf16)sHw[rr * 16 + ln];          // rows 26..31 are zero
        }
        return __builtin_amdgcn_wmma_f32_16x16x32_bf16(
            false, Aw, false, Bh, (short)0, z, false, false);
    };

    v8f mu1 = conv_field(Ax);
    v8f mu2 = conv_field(Ay);
    v8f gxx = conv_field(Axx);
    v8f gyy = conv_field(Ayy);
    v8f gxy = conv_field(Axy);

    // ---- SSIM / CS maps + masked accumulation (C/D layout: M=r+8*half, N=ln)
    float ssum = 0.0f, csum = 0.0f;
    const int ocol = colBase + wcol0 + ln;
#pragma unroll
    for (int r = 0; r < 8; ++r) {
        int orow = rowBase + r + 8 * half;
        if (orow < Hv && ocol < Wv) {
            float m1 = mu1[r], m2 = mu2[r];
            float m1s = m1 * m1, m2s = m2 * m2, m12 = m1 * m2;
            float s1  = gxx[r] - m1s;
            float s2  = gyy[r] - m2s;
            float s12 = gxy[r] - m12;
            float cs  = (2.0f * s12 + C2_) / (s1 + s2 + C2_);
            float ss  = ((2.0f * m12 + C1_) / (m1s + m2s + C1_)) * cs;
            csum += cs;
            ssum += ss;
        }
    }

    // ---- wave32 reduction, one pair of global f32 atomics per wave ----
#pragma unroll
    for (int off = 16; off > 0; off >>= 1) {
        ssum += __shfl_down(ssum, off, 32);
        csum += __shfl_down(csum, off, 32);
    }
    if (lane == 0) {
        atomicAdd(&accSsim[nc], ssum);
        atomicAdd(&accCs[nc], csum);
    }
}

// ---------------------------------------------------------------------------
// 2x2 average pool, stride 2 (dims always even here: 512->256->128->64->32)
// ---------------------------------------------------------------------------
__global__ void pool2_kernel(const float* __restrict__ in, float* __restrict__ out,
                             int Hi, int Wi, long nOut)
{
    long i = (long)blockIdx.x * blockDim.x + threadIdx.x;
    if (i >= nOut) return;
    int Ho = Hi >> 1, Wo = Wi >> 1;
    long pl  = (long)Ho * Wo;
    long nc  = i / pl;
    long rem = i - nc * pl;
    int h = (int)(rem / Wo), w = (int)(rem % Wo);
    const float* p = in + nc * (long)Hi * Wi + (long)(2 * h) * Wi + 2 * w;
    out[i] = 0.25f * (p[0] + p[1] + p[Wi] + p[Wi + 1]);
}

// ---------------------------------------------------------------------------
// Final combine: ms = prod_l relu(val_l)^w_l averaged over (n,c); out = 1-mean
// acc layout: [level][{ssim,cs}][96]
// ---------------------------------------------------------------------------
__global__ void combine_kernel(const float* __restrict__ acc, float* __restrict__ out)
{
    __shared__ float red[96];
    const float wts[5]  = {0.0448f, 0.2856f, 0.3001f, 0.2363f, 0.1333f};
    const float cnts[5] = {502.0f * 502.0f, 246.0f * 246.0f, 118.0f * 118.0f,
                           54.0f * 54.0f,   22.0f * 22.0f};
    int t = threadIdx.x;
    if (t < 96) {
        float ms = 1.0f;
#pragma unroll
        for (int l = 0; l < 5; ++l) {
            float v = (l < 4) ? acc[(l * 2 + 1) * 96 + t]   // cs  (levels 0..3)
                              : acc[(l * 2 + 0) * 96 + t];  // ssim (level 4)
            v = fmaxf(v / cnts[l], 0.0f);
            ms *= powf(v, wts[l]);
        }
        red[t] = ms;
    }
    __syncthreads();
    if (t == 0) {
        float s = 0.0f;
        for (int i = 0; i < 96; ++i) s += red[i];
        out[0] = 1.0f - s / 96.0f;
    }
}

// ---------------------------------------------------------------------------
extern "C" void kernel_launch(void* const* d_in, const int* in_sizes, int n_in,
                              void* d_out, int out_size, void* d_ws, size_t ws_size,
                              hipStream_t stream)
{
    (void)in_sizes; (void)n_in; (void)out_size; (void)ws_size;
    const float* X0 = (const float*)d_in[0];
    const float* Y0 = (const float*)d_in[1];

    const int NC = 32 * 3;
    static const int HS[5] = {512, 256, 128, 64, 32};

    // workspace: pyramid levels 1..4 for X and Y, then accumulators
    float* ws = (float*)d_ws;
    size_t off = 0;
    float* Xl[5] = {nullptr}; float* Yl[5] = {nullptr};
    for (int l = 1; l < 5; ++l) {
        size_t n = (size_t)NC * HS[l] * HS[l];
        Xl[l] = ws + off; off += n;
        Yl[l] = ws + off; off += n;
    }
    float* acc = ws + off;                 // 5 * 2 * 96 floats
    hipMemsetAsync(acc, 0, 5 * 2 * 96 * sizeof(float), stream);

    const float* Xc = X0;
    const float* Yc = Y0;
    for (int l = 0; l < 5; ++l) {
        int H = HS[l], W = HS[l];
        int Hv = H - 10, Wv = W - 10;
        dim3 grid((Wv + 63) / 64, (Hv + 15) / 16, NC);
        ssim_stats_kernel<<<grid, 128, 0, stream>>>(
            Xc, Yc, acc + (l * 2 + 0) * 96, acc + (l * 2 + 1) * 96, H, W, Hv, Wv);
        if (l < 4) {
            long nOut = (long)NC * HS[l + 1] * HS[l + 1];
            int  thr  = 256;
            unsigned blks = (unsigned)((nOut + thr - 1) / thr);
            pool2_kernel<<<blks, thr, 0, stream>>>(Xc, Xl[l + 1], H, W, nOut);
            pool2_kernel<<<blks, thr, 0, stream>>>(Yc, Yl[l + 1], H, W, nOut);
            Xc = Xl[l + 1];
            Yc = Yl[l + 1];
        }
    }
    combine_kernel<<<1, 128, 0, stream>>>(acc, (float*)d_out);
}
